// BiDACPI_69011534512277
// MI455X (gfx1250) — matrix operand order repaired and checked by reading, not compiled, not run on hardware
//
#include <hip/hip_runtime.h>
#include <hip/hip_bf16.h>

// ---------------------------------------------------------------------------
// Problem constants (match reference)
// ---------------------------------------------------------------------------
#define CB   32      // batch
#define NN_  256     // atoms
#define LL   2000    // amino length
#define CC   128     // channels
#define GG   64      // GAT head dim
#define HH   4       // heads
#define DMAI 4       // DMA iterations

#define ACT_NONE 0
#define ACT_LEAKY 1
#define ACT_TANH 2
#define ACT_SIGMOID 3
#define ACT_ELU 4

typedef __bf16 v16bf __attribute__((ext_vector_type(16)));
typedef float v8f __attribute__((ext_vector_type(8)));

static __device__ __forceinline__ float actf(float x, int act) {
  switch (act) {
    case ACT_LEAKY:   return x >= 0.f ? x : 0.1f * x;
    case ACT_TANH:    return tanhf(x);
    case ACT_SIGMOID: return 1.f / (1.f + expf(-x));
    case ACT_ELU:     return x > 0.f ? x : expf(x) - 1.f;
    default:          return x;
  }
}

static __device__ __forceinline__ v8f wmma_bf16(v16bf a, v16bf b, v8f c) {
  return __builtin_amdgcn_wmma_f32_16x16x32_bf16(false, a, false, b,
                                                 (short)0, c, false, false);
}

// ---------------------------------------------------------------------------
// Fragment loaders. k order used by this code (contiguous, not the ISA table order):
//   A 16x32 bf16 : lane = M (0..15); K = laneHi*16 + j (j = 0..15),
//                  element j=2v+h.
//   B 32x16 bf16 : lane = N (0..15); K = laneHi*16 + j  (j = 0..15).
// GUARD=true zero-pads K tail; TA/TB select (m,k)/(k,n) vs transposed storage.
// Conversion fp32->bf16 via native cast (compiler emits v_cvt_*_bf16).
// ---------------------------------------------------------------------------
template<bool GUARD, int TA>
static __device__ __forceinline__ v16bf load_afrag(const float* __restrict__ A, int lda,
                                                   int mrow, int k0, int K, int koffA) {
  v16bf au;
  if (TA == 0) {
    const float* ap = A + (long long)mrow * lda + k0;
#pragma unroll
    for (int v = 0; v < 8; ++v) {
      const int kb = koffA + 2 * v;
      if (!GUARD) {
        float2 xy = *(const float2*)(ap + kb);       // global_load_b64
        au[2 * v]     = (__bf16)xy.x;
        au[2 * v + 1] = (__bf16)xy.y;
      } else {
        float x0 = (k0 + kb     < K) ? ap[kb]     : 0.f;
        float x1 = (k0 + kb + 1 < K) ? ap[kb + 1] : 0.f;
        au[2 * v]     = (__bf16)x0;
        au[2 * v + 1] = (__bf16)x1;
      }
    }
  } else {  // A stored K-major: element (m,k) at A[k*lda + m]
#pragma unroll
    for (int v = 0; v < 8; ++v) {
      const int kb = koffA + 2 * v;
#pragma unroll
      for (int h = 0; h < 2; ++h) {
        const int ka = k0 + kb + h;
        float x = (!GUARD || ka < K) ? A[(long long)ka * lda + mrow] : 0.f;
        au[2 * v + h] = (__bf16)x;
      }
    }
  }
  return au;
}

template<bool GUARD, int TB>
static __device__ __forceinline__ v16bf load_bfrag(const float* __restrict__ Bm, int ldb,
                                                   int ncol, int k0, int K, int koffB) {
  v16bf bu;
  if (TB == 0) {  // B stored (K x N): strided along K
    const float* bp = Bm + (long long)(k0 + koffB) * ldb + ncol;
#pragma unroll
    for (int j = 0; j < 16; ++j) {
      float y = (!GUARD || (k0 + koffB + j) < K) ? bp[(long long)j * ldb] : 0.f;
      bu[j] = (__bf16)y;
    }
  } else {        // B stored (N x K): 16 contiguous K values
    const float* bp = Bm + (long long)ncol * ldb + k0 + koffB;
    if (!GUARD) {
#pragma unroll
      for (int q = 0; q < 4; ++q) {
        float4 f = *(const float4*)(bp + 4 * q);     // global_load_b128
        bu[4 * q]     = (__bf16)f.x;
        bu[4 * q + 1] = (__bf16)f.y;
        bu[4 * q + 2] = (__bf16)f.z;
        bu[4 * q + 3] = (__bf16)f.w;
      }
    } else {
#pragma unroll
      for (int j = 0; j < 16; ++j) {
        float y = (k0 + koffB + j < K) ? bp[j] : 0.f;
        bu[j] = (__bf16)y;
      }
    }
  }
  return bu;
}

// ---------------------------------------------------------------------------
// Loop + epilogue body, specialized on HAS2 so the second-tile work is
// branch-free inside the K loop (no per-iteration exec juggling).
// ---------------------------------------------------------------------------
template<int TA, int TB, bool HAS2>
static __device__ __forceinline__ void gemm_body(
    const float* __restrict__ A, int lda,
    const float* __restrict__ Bm, int ldb,
    float* __restrict__ C, int ldc,
    const float* __restrict__ bias, const float* __restrict__ rowMask,
    const float* __restrict__ colMask,
    int M, int N, int K, int act, int b, int mt,
    int mrow, int nc0, int koffA, int koffB, int lane) {
  v8f acc0 = {0.f, 0.f, 0.f, 0.f, 0.f, 0.f, 0.f, 0.f};
  v8f acc1 = {0.f, 0.f, 0.f, 0.f, 0.f, 0.f, 0.f, 0.f};

  const int Kfull = K & ~31;
  int k0 = 0;
#pragma unroll 2
  for (; k0 < Kfull; k0 += 32) {          // unguarded hot loop
    v16bf af  = load_afrag<false, TA>(A, lda, mrow, k0, K, koffA);
    v16bf bf0 = load_bfrag<false, TB>(Bm, ldb, nc0, k0, K, koffB);
    acc0 = wmma_bf16(af, bf0, acc0);
    if (HAS2) {
      v16bf bf1 = load_bfrag<false, TB>(Bm, ldb, nc0 + 16, k0, K, koffB);
      acc1 = wmma_bf16(af, bf1, acc1);
    }
  }
  if (k0 < K) {                           // guarded zero-padded tail
    v16bf af  = load_afrag<true, TA>(A, lda, mrow, k0, K, koffA);
    v16bf bf0 = load_bfrag<true, TB>(Bm, ldb, nc0, k0, K, koffB);
    acc0 = wmma_bf16(af, bf0, acc0);
    if (HAS2) {
      v16bf bf1 = load_bfrag<true, TB>(Bm, ldb, nc0 + 16, k0, K, koffB);
      acc1 = wmma_bf16(af, bf1, acc1);
    }
  }

  const int rbase = mt + ((lane >> 4) << 3);
#pragma unroll
  for (int r = 0; r < 8; ++r) {
    const int row = rbase + r;
    float x0 = acc0[r];
    if (bias)    x0 += bias[nc0];
    x0 = actf(x0, act);
    if (rowMask) x0 *= rowMask[(long long)b * M + row];
    if (colMask) x0 *= colMask[(long long)b * N + nc0];
    C[(long long)row * ldc + nc0] = x0;
    if (HAS2) {
      float x1 = acc1[r];
      if (bias)    x1 += bias[nc0 + 16];
      x1 = actf(x1, act);
      if (rowMask) x1 *= rowMask[(long long)b * M + row];
      if (colMask) x1 *= colMask[(long long)b * N + nc0 + 16];
      C[(long long)row * ldc + nc0 + 16] = x1;
    }
  }
}

// ---------------------------------------------------------------------------
// Generic batched GEMM on bf16 WMMA, fp32 accumulate.
//   C[b] (MxN) = actmask( A[b] (MxK) @ B[b] (KxN) + bias )
// Block = 128 threads = 4 waves; wave tile = 16x32 (2 accumulators sharing
// one A fragment); block covers 32 rows x 64 cols. M,N multiples of 16;
// K arbitrary (zero-padded tail fragment).
// ---------------------------------------------------------------------------
template<int TA, int TB>
__global__ __launch_bounds__(128) void wmma_gemm_t(
    const float* __restrict__ A, long long bsA, int lda,
    const float* __restrict__ Bm, long long bsB, int ldb,
    float* __restrict__ C, long long bsC, int ldc,
    const float* __restrict__ bias,
    const float* __restrict__ rowMask,
    const float* __restrict__ colMask,
    int M, int N, int K, int act) {
  const int lane = threadIdx.x & 31;
  const int wid  = threadIdx.x >> 5;
  const int b    = blockIdx.z;
  const int mt   = blockIdx.y * 32 + (wid >> 1) * 16;
  const int nt0  = blockIdx.x * 64 + (wid & 1) * 32;
  if (mt >= M || nt0 >= N) return;           // wave-uniform, EXEC stays full

  A  += (long long)b * bsA;
  Bm += (long long)b * bsB;
  C  += (long long)b * bsC;

  const int mrow  = mt  + (lane & 15);
  const int nc0   = nt0 + (lane & 15);
  const int koffA = (lane >> 4) << 4;   // 0 or 16
  const int koffB = (lane >> 4) << 4;   // 0 or 16

  if ((nt0 + 16) < N)
    gemm_body<TA, TB, true >(A, lda, Bm, ldb, C, ldc, bias, rowMask, colMask,
                             M, N, K, act, b, mt, mrow, nc0, koffA, koffB, lane);
  else
    gemm_body<TA, TB, false>(A, lda, Bm, ldb, C, ldc, bias, rowMask, colMask,
                             M, N, K, act, b, mt, mrow, nc0, koffA, koffB, lane);
}

// ---------------------------------------------------------------------------
// Small helper kernels
// ---------------------------------------------------------------------------
__global__ void embed_rows(const int* __restrict__ idx, const float* __restrict__ emb,
                           float* __restrict__ out, long long rows, int C) {
  long long i = (long long)blockIdx.x * blockDim.x + threadIdx.x;
  if (i >= rows * (long long)C) return;
  long long r = i / C; int c = (int)(i % C);
  out[i] = emb[(long long)idx[r] * C + c];
}

// s1[r]=dot(Wh[r,:],a[:G]); s2[r]=dot(Wh[r,:],a[G:])
__global__ void gat_scores(const float* __restrict__ Wh, const float* __restrict__ a,
                           int G, long long rows, float* __restrict__ ssrc,
                           float* __restrict__ sdst) {
  long long r = (long long)blockIdx.x * blockDim.x + threadIdx.x;
  if (r >= rows) return;
  const float* w = Wh + r * (long long)G;
  float s1 = 0.f, s2 = 0.f;
  for (int g = 0; g < G; ++g) { s1 += w[g] * a[g]; s2 += w[g] * a[G + g]; }
  ssrc[r] = s1; sdst[r] = s2;
}

// att[b,r,:] = softmax_m( mask(adj, leaky(ssrc[b,r]+sdst[b,m])) ), n==blockDim==256
__global__ void gat_att(const float* __restrict__ ssrc, const float* __restrict__ sdst,
                        const float* __restrict__ adj, float* __restrict__ att, int n) {
  __shared__ float red[256];
  const int b = blockIdx.y, r = blockIdx.x, t = threadIdx.x;
  float e = ssrc[(long long)b * n + r] + sdst[(long long)b * n + t];
  e = e >= 0.f ? e : 0.1f * e;
  float a = adj[((long long)b * n + r) * n + t];
  e = (a > 0.f) ? e : -9.0e15f;
  red[t] = e; __syncthreads();
  for (int s = 128; s > 0; s >>= 1) { if (t < s) red[t] = fmaxf(red[t], red[t + s]); __syncthreads(); }
  float mx = red[0]; __syncthreads();
  float ex = expf(e - mx);
  red[t] = ex; __syncthreads();
  for (int s = 128; s > 0; s >>= 1) { if (t < s) red[t] += red[t + s]; __syncthreads(); }
  att[((long long)b * n + r) * n + t] = ex / red[0];
}

// 11x11 "same" conv over (L,C) plane, single channel, + bias + leaky
__global__ void conv11(const float* __restrict__ in, const float* __restrict__ kern,
                       const float* __restrict__ bias, float* __restrict__ out,
                       int Lr, int Cc) {
  __shared__ float tile[26][27];
  __shared__ float kk[121];
  const int b = blockIdx.z;
  const int y0 = blockIdx.y * 16, x0 = blockIdx.x * 16;
  const int tx = threadIdx.x, ty = threadIdx.y;
  const int tid = ty * 16 + tx;
  if (tid < 121) kk[tid] = kern[tid];
  const float* inb = in + (long long)b * Lr * Cc;
  for (int idx = tid; idx < 26 * 26; idx += 256) {
    int yy = idx / 26, xx = idx % 26;
    int gy = y0 + yy - 5, gx = x0 + xx - 5;
    float v = 0.f;
    if (gy >= 0 && gy < Lr && gx >= 0 && gx < Cc) v = inb[(long long)gy * Cc + gx];
    tile[yy][xx] = v;
  }
  __syncthreads();
  int gy = y0 + ty, gx = x0 + tx;
  if (gy < Lr && gx < Cc) {
    float acc = bias[0];
#pragma unroll
    for (int dy = 0; dy < 11; ++dy)
#pragma unroll
      for (int dx = 0; dx < 11; ++dx)
        acc += kk[dy * 11 + dx] * tile[ty + dy][tx + dx];
    acc = acc >= 0.f ? acc : 0.1f * acc;
    out[(long long)b * Lr * Cc + (long long)gy * Cc + gx] = acc;
  }
}

// out[b,c] = sum_r feat[b,r,c]*mask[b,r] / sum_r mask[b,r]   (blockDim = C)
__global__ void masked_mean(const float* __restrict__ feat, const float* __restrict__ mask,
                            float* __restrict__ out, int n, int C) {
  const int b = blockIdx.x, c = threadIdx.x;
  float s = 0.f, ms = 0.f;
  for (int r = 0; r < n; ++r) {
    float mk = mask[(long long)b * n + r];
    s  += feat[((long long)b * n + r) * C + c] * mk;
    ms += mk;
  }
  out[b * C + c] = s / ms;
}

__global__ void ewise_mul(const float* __restrict__ x, const float* __restrict__ y,
                          float* __restrict__ z, long long n) {
  long long i = (long long)blockIdx.x * blockDim.x + threadIdx.x;
  if (i < n) z[i] = x[i] * y[i];
}

// h[b,r,c] *= mg[b,c] * other[b,r,c]
__global__ void gate_mul(float* __restrict__ h, const float* __restrict__ mg,
                         const float* __restrict__ other, int rowsPerB, int C,
                         long long ntot) {
  long long i = (long long)blockIdx.x * blockDim.x + threadIdx.x;
  if (i >= ntot) return;
  long long b = i / ((long long)rowsPerB * C);
  int c = (int)(i % C);
  h[i] = h[i] * mg[b * C + c] * other[i];
}

// out[r] = b0[0] + dot(X[r,:], w)
__global__ void vec_dot_rows(const float* __restrict__ X, const float* __restrict__ w,
                             const float* __restrict__ b0, float* __restrict__ out,
                             long long rows, int C) {
  long long r = (long long)blockIdx.x * blockDim.x + threadIdx.x;
  if (r >= rows) return;
  const float* xr = X + r * (long long)C;
  float acc = b0[0];
  for (int c = 0; c < C; ++c) acc += xr[c] * w[c];
  out[r] = acc;
}

// att = exp(s - max_all(s)) * mask / (sum + 1e-6)   (blockDim = 256)
__global__ void mask_softmax_k(const float* __restrict__ s, const float* __restrict__ mask,
                               float* __restrict__ att, int n) {
  __shared__ float red[256];
  const int b = blockIdx.x, t = threadIdx.x;
  const float* sb = s + (long long)b * n;
  const float* mb = mask + (long long)b * n;
  float mx = -3.4e38f;
  for (int i = t; i < n; i += 256) mx = fmaxf(mx, sb[i]);
  red[t] = mx; __syncthreads();
  for (int st = 128; st > 0; st >>= 1) { if (t < st) red[t] = fmaxf(red[t], red[t + st]); __syncthreads(); }
  mx = red[0]; __syncthreads();
  float sum = 0.f;
  for (int i = t; i < n; i += 256) sum += expf(sb[i] - mx) * mb[i];
  red[t] = sum; __syncthreads();
  for (int st = 128; st > 0; st >>= 1) { if (t < st) red[t] += red[t + st]; __syncthreads(); }
  sum = red[0] + 1e-6f;
  for (int i = t; i < n; i += 256) att[(long long)b * n + i] = expf(sb[i] - mx) * mb[i] / sum;
}

// out[b,c] = sum_r feat[b,r,c]*att[b,r]   (blockDim = C)
__global__ void att_reduce(const float* __restrict__ feat, const float* __restrict__ att,
                           float* __restrict__ out, int n, int C) {
  const int b = blockIdx.x, c = threadIdx.x;
  float s = 0.f;
  for (int r = 0; r < n; ++r)
    s += feat[((long long)b * n + r) * C + c] * att[(long long)b * n + r];
  out[b * C + c] = s;
}

// GRU cell update: m = (1-z)*n + z*m, blockDim = 3C = 384
__global__ void gru_step(const float* __restrict__ cf, const float* __restrict__ pf,
                         float* __restrict__ m, const float* __restrict__ wih,
                         const float* __restrict__ whh, const float* __restrict__ bih,
                         const float* __restrict__ bhh, int C) {
  __shared__ float x[128], h[128], gi[384], gh[384];
  const int b = blockIdx.x, t = threadIdx.x;
  if (t < C) { x[t] = cf[b * C + t] * pf[b * C + t]; h[t] = m[b * C + t]; }
  __syncthreads();
  float si = bih[t], sh = bhh[t];
  const float* wi = wih + (long long)t * C;
  const float* wh = whh + (long long)t * C;
  for (int c = 0; c < C; ++c) { si += x[c] * wi[c]; sh += h[c] * wh[c]; }
  gi[t] = si; gh[t] = sh;
  __syncthreads();
  if (t < C) {
    float r = 1.f / (1.f + expf(-(gi[t] + gh[t])));
    float z = 1.f / (1.f + expf(-(gi[C + t] + gh[C + t])));
    float nn = tanhf(gi[2 * C + t] + r * gh[2 * C + t]);
    m[b * C + t] = (1.f - z) * nn + z * h[t];
  }
}

// out[b] = bout[0] + sum_{i<256,j<128} leaky(cfcat[b,i]*pf[b,j]) * Wout[i*128+j]
__global__ void final_out(const float* __restrict__ cf, const float* __restrict__ sfin,
                          const float* __restrict__ pf, const float* __restrict__ Wout,
                          const float* __restrict__ bout, float* __restrict__ out) {
  __shared__ float red[256];
  const int b = blockIdx.x, t = threadIdx.x;
  float ci = (t < 128) ? cf[b * 128 + t] : sfin[b * 128 + t - 128];
  const float* pfb = pf + b * 128;
  const float* wr = Wout + (long long)t * 128;
  float acc = 0.f;
  for (int j = 0; j < 128; ++j) {
    float v = ci * pfb[j];
    v = v >= 0.f ? v : 0.1f * v;
    acc += v * wr[j];
  }
  red[t] = acc; __syncthreads();
  for (int s = 128; s > 0; s >>= 1) { if (t < s) red[t] += red[t + s]; __syncthreads(); }
  if (t == 0) out[b] = red[0] + bout[0];
}

// ---------------------------------------------------------------------------
// Host orchestration
// ---------------------------------------------------------------------------
extern "C" void kernel_launch(void* const* d_in, const int* in_sizes, int n_in,
                              void* d_out, int out_size, void* d_ws, size_t ws_size,
                              hipStream_t stream) {
  const int*   atoms      = (const int*)  d_in[0];
  const float* atoms_mask = (const float*)d_in[1];
  const float* adj        = (const float*)d_in[2];
  const int*   amino      = (const int*)  d_in[3];
  const float* amino_mask = (const float*)d_in[4];
  const float* fps        = (const float*)d_in[5];
  const float* emb_atom   = (const float*)d_in[6];
  const float* emb_amino  = (const float*)d_in[7];
  const float* gatW       = (const float*)d_in[8];
  const float* gata       = (const float*)d_in[9];
  const float* goW        = (const float*)d_in[10];
  const float* goa        = (const float*)d_in[11];
  const float* WcompW     = (const float*)d_in[12];
  const float* Wcompb     = (const float*)d_in[13];
  const float* convK      = (const float*)d_in[14];
  const float* convb      = (const float*)d_in[15];
  const float* WprotW     = (const float*)d_in[16];
  const float* Wprotb     = (const float*)d_in[17];
  const float* pcW        = (const float*)d_in[18];
  const float* pcb        = (const float*)d_in[19];
  const float* ppW        = (const float*)d_in[20];
  const float* ppb        = (const float*)d_in[21];
  const float* cfW        = (const float*)d_in[22];
  const float* cfb        = (const float*)d_in[23];
  const float* pfW        = (const float*)d_in[24];
  const float* pfb        = (const float*)d_in[25];
  const float* sfW        = (const float*)d_in[26];
  const float* sfb        = (const float*)d_in[27];
  const float* fp0        = (const float*)d_in[28];
  const float* fp1        = (const float*)d_in[29];
  const float* c2pW       = (const float*)d_in[30];
  const float* c2pb       = (const float*)d_in[31];
  const float* p2cW       = (const float*)d_in[32];
  const float* p2cb       = (const float*)d_in[33];
  const float* mc1W       = (const float*)d_in[34];
  const float* mc1b       = (const float*)d_in[35];
  const float* mp1W       = (const float*)d_in[36];
  const float* mp1b       = (const float*)d_in[37];
  const float* hc0W       = (const float*)d_in[38];
  const float* hc0b       = (const float*)d_in[39];
  const float* hp0W       = (const float*)d_in[40];
  const float* hp0b       = (const float*)d_in[41];
  const float* hc1W       = (const float*)d_in[42];
  const float* hc1b       = (const float*)d_in[43];
  const float* hp1W       = (const float*)d_in[44];
  const float* hp1b       = (const float*)d_in[45];
  const float* gru_wih    = (const float*)d_in[46];
  const float* gru_whh    = (const float*)d_in[47];
  const float* gru_bih    = (const float*)d_in[48];
  const float* gru_bhh    = (const float*)d_in[49];
  const float* WoutW      = (const float*)d_in[50];
  const float* Woutb      = (const float*)d_in[51];
  (void)in_sizes; (void)n_in; (void)out_size; (void)ws_size;

  // -------- workspace layout (floats), lifetime-based aliasing --------
  float* W = (float*)d_ws;
  const long long BIG = (long long)CB * LL * CC;      // 8,192,000
  float* pvA   = W;                   // pv; reused as hph in the DMA loop
  float* pvB   = W + BIG;             // conv ping; reused as c2p
  float* r_pp  = W + 2 * BIG;         // pp; reused as tp
  float* pfeat = W + 3 * BIG;
  float* pair  = W + 4 * BIG;         // B*N*L
  long long off = 4 * BIG + (long long)CB * NN_ * LL;
  const long long SZ_BNC = (long long)CB * NN_ * CC;  // 1,048,576
  float* av0   = W + off; off += SZ_BNC;
  float* Wh    = W + off; off += SZ_BNC;              // fits 8192x128 and 8192x64
  float* attb  = W + off; off += (long long)CB * NN_ * NN_;
  float* multi = W + off; off += (long long)CB * NN_ * (HH * GG);
  float* av1   = W + off; off += SZ_BNC;
  float* av2   = W + off; off += SZ_BNC;
  float* r_pc  = W + off; off += SZ_BNC;              // pc; reused as tc
  float* cfeat = W + off; off += SZ_BNC;
  float* p2c   = W + off; off += SZ_BNC;
  float* hch   = W + off; off += SZ_BNC;
  float* ssrc  = W + off; off += (long long)CB * LL;  // also cs / ps scores
  float* sdst  = W + off; off += (long long)CB * LL;
  float* catt  = W + off; off += (long long)CB * NN_;
  float* patt  = W + off; off += (long long)CB * LL;
  float* cf    = W + off; off += CB * CC;
  float* pf    = W + off; off += CB * CC;
  float* mm    = W + off; off += CB * CC;
  float* mcv   = W + off; off += CB * CC;
  float* mpv   = W + off; off += CB * CC;
  float* sf0   = W + off; off += CB * CC;
  float* sf1   = W + off; off += CB * CC;
  float* sfin  = W + off; off += CB * CC;
  float* tc  = r_pc;   // aliases (dead producers)
  float* tp  = r_pp;
  float* c2p = pvB;
  float* hph = pvA;

  auto gemm = [&](const float* A, long long bsA, int lda, int tA,
                  const float* Bm, long long bsB, int ldb, int tB,
                  float* Cm, long long bsC, int ldc,
                  const float* bias, const float* rm, const float* cm,
                  int M, int N, int K, int act, int batch) {
    dim3 g((N + 63) / 64, (M + 31) / 32, batch);
    if (!tA && !tB)
      wmma_gemm_t<0, 0><<<g, dim3(128), 0, stream>>>(A, bsA, lda, Bm, bsB, ldb,
                                                     Cm, bsC, ldc, bias, rm, cm, M, N, K, act);
    else if (!tA && tB)
      wmma_gemm_t<0, 1><<<g, dim3(128), 0, stream>>>(A, bsA, lda, Bm, bsB, ldb,
                                                     Cm, bsC, ldc, bias, rm, cm, M, N, K, act);
    else
      wmma_gemm_t<1, 0><<<g, dim3(128), 0, stream>>>(A, bsA, lda, Bm, bsB, ldb,
                                                     Cm, bsC, ldc, bias, rm, cm, M, N, K, act);
  };

  // ---------------- ligand path: embeddings + GAT ----------------
  embed_rows<<<(unsigned)((SZ_BNC + 255) / 256), 256, 0, stream>>>(atoms, emb_atom, av0, (long long)CB * NN_, CC);

  for (int i = 0; i < HH; ++i) {
    gemm(av0, 0, CC, 0, gatW + (long long)i * CC * GG, 0, GG, 0,
         Wh, 0, GG, nullptr, nullptr, nullptr, CB * NN_, GG, CC, ACT_NONE, 1);
    gat_scores<<<(CB * NN_ + 255) / 256, 256, 0, stream>>>(Wh, gata + (long long)i * 2 * GG, GG, (long long)CB * NN_, ssrc, sdst);
    gat_att<<<dim3(NN_, CB), NN_, 0, stream>>>(ssrc, sdst, adj, attb, NN_);
    gemm(attb, (long long)NN_ * NN_, NN_, 0, Wh, (long long)NN_ * GG, GG, 0,
         multi + i * GG, (long long)NN_ * (HH * GG), HH * GG,
         nullptr, nullptr, nullptr, NN_, GG, NN_, ACT_ELU, CB);
  }
  // output GAT (concat=False, elu applied outside)
  gemm(multi, 0, HH * GG, 0, goW, 0, CC, 0, Wh, 0, CC,
       nullptr, nullptr, nullptr, CB * NN_, CC, HH * GG, ACT_NONE, 1);
  gat_scores<<<(CB * NN_ + 255) / 256, 256, 0, stream>>>(Wh, goa, CC, (long long)CB * NN_, ssrc, sdst);
  gat_att<<<dim3(NN_, CB), NN_, 0, stream>>>(ssrc, sdst, adj, attb, NN_);
  gemm(attb, (long long)NN_ * NN_, NN_, 0, Wh, (long long)NN_ * CC, CC, 0,
       av1, (long long)NN_ * CC, CC, nullptr, nullptr, nullptr, NN_, CC, NN_, ACT_ELU, CB);
  // Wcomp + leaky + atom mask
  gemm(av1, 0, CC, 0, WcompW, 0, CC, 0, av2, 0, CC,
       Wcompb, atoms_mask, nullptr, CB * NN_, CC, CC, ACT_LEAKY, 1);

  // ---------------- protein path: embed + 3x conv11 + Wprot ----------------
  embed_rows<<<(unsigned)((BIG + 255) / 256), 256, 0, stream>>>(amino, emb_amino, pvA, (long long)CB * LL, CC);
  conv11<<<dim3(CC / 16, LL / 16, CB), dim3(16, 16), 0, stream>>>(pvA, convK + 0,   convb + 0, pvB, LL, CC);
  conv11<<<dim3(CC / 16, LL / 16, CB), dim3(16, 16), 0, stream>>>(pvB, convK + 121, convb + 1, pvA, LL, CC);
  conv11<<<dim3(CC / 16, LL / 16, CB), dim3(16, 16), 0, stream>>>(pvA, convK + 242, convb + 2, pvB, LL, CC);
  gemm(pvB, 0, CC, 0, WprotW, 0, CC, 0, pvA, 0, CC,
       Wprotb, amino_mask, nullptr, CB * LL, CC, CC, ACT_LEAKY, 1);

  // ---------------- fingerprint path ----------------
  gemm(fps, 0, 1024, 0, fp0, 0, CC, 0, sf0, 0, CC, nullptr, nullptr, nullptr, CB, CC, 1024, ACT_LEAKY, 1);
  gemm(sf0, 0, CC, 0, fp1, 0, CC, 0, sf1, 0, CC, nullptr, nullptr, nullptr, CB, CC, CC, ACT_LEAKY, 1);
  gemm(sf1, 0, CC, 0, sfW, 0, CC, 0, sfin, 0, CC, sfb, nullptr, nullptr, CB, CC, CC, ACT_LEAKY, 1);

  // ---------------- pair matrix + features ----------------
  gemm(av2, 0, CC, 0, pcW, 0, CC, 0, r_pc, 0, CC, pcb, nullptr, nullptr, CB * NN_, CC, CC, ACT_LEAKY, 1);
  gemm(pvA, 0, CC, 0, ppW, 0, CC, 0, r_pp, 0, CC, ppb, nullptr, nullptr, CB * LL, CC, CC, ACT_LEAKY, 1);
  gemm(av2, 0, CC, 0, cfW, 0, CC, 0, cfeat, 0, CC, cfb, nullptr, nullptr, CB * NN_, CC, CC, ACT_LEAKY, 1);
  gemm(pvA, 0, CC, 0, pfW, 0, CC, 0, pfeat, 0, CC, pfb, nullptr, nullptr, CB * LL, CC, CC, ACT_LEAKY, 1);
  // pair[b] = sigmoid(pc[b] @ pp[b]^T) * rowmask * colmask
  gemm(r_pc, (long long)NN_ * CC, CC, 0, r_pp, (long long)LL * CC, CC, 1,
       pair, (long long)NN_ * LL, LL, nullptr, atoms_mask, amino_mask,
       NN_, LL, CC, ACT_SIGMOID, CB);

  // ---------------- initial memory state ----------------
  masked_mean<<<CB, CC, 0, stream>>>(cfeat, atoms_mask, cf, NN_, CC);
  masked_mean<<<CB, CC, 0, stream>>>(pfeat, amino_mask, pf, LL, CC);
  ewise_mul<<<(CB * CC + 255) / 256, 256, 0, stream>>>(cf, pf, mm, (long long)CB * CC);

  // ---------------- DMA loop ----------------
  for (int i = 0; i < DMAI; ++i) {
    const long long wofs = (long long)i * CC * CC;
    // tc = tanh(cfeat @ c2pW[i] + c2pb[i]);  c2p[b] = pair[b]^T @ tc[b]
    gemm(cfeat, 0, CC, 0, c2pW + wofs, 0, CC, 0, tc, 0, CC,
         c2pb + i * CC, nullptr, nullptr, CB * NN_, CC, CC, ACT_TANH, 1);
    gemm(pair, (long long)NN_ * LL, LL, 1, tc, (long long)NN_ * CC, CC, 0,
         c2p, (long long)LL * CC, CC, nullptr, nullptr, nullptr, LL, CC, NN_, ACT_NONE, CB);
    // tp = tanh(pfeat @ p2cW[i] + p2cb[i]);  p2c[b] = pair[b] @ tp[b]
    gemm(pfeat, 0, CC, 0, p2cW + wofs, 0, CC, 0, tp, 0, CC,
         p2cb + i * CC, nullptr, nullptr, CB * LL, CC, CC, ACT_TANH, 1);
    gemm(pair, (long long)NN_ * LL, LL, 0, tp, (long long)LL * CC, CC, 0,
         p2c, (long long)NN_ * CC, CC, nullptr, nullptr, nullptr, NN_, CC, LL, ACT_NONE, CB);
    // ctmp = tanh(cfeat@hc0W[i]+b) * tanh(m@mc1W[i]+b)[:,None,:] * p2c
    gemm(cfeat, 0, CC, 0, hc0W + wofs, 0, CC, 0, hch, 0, CC,
         hc0b + i * CC, nullptr, nullptr, CB * NN_, CC, CC, ACT_TANH, 1);
    gemm(mm, 0, CC, 0, mc1W + wofs, 0, CC, 0, mcv, 0, CC,
         mc1b + i * CC, nullptr, nullptr, CB, CC, CC, ACT_TANH, 1);
    gate_mul<<<(unsigned)((SZ_BNC + 255) / 256), 256, 0, stream>>>(hch, mcv, p2c, NN_, CC, SZ_BNC);
    // ptmp = tanh(pfeat@hp0W[i]+b) * tanh(m@mp1W[i]+b)[:,None,:] * c2p
    gemm(pfeat, 0, CC, 0, hp0W + wofs, 0, CC, 0, hph, 0, CC,
         hp0b + i * CC, nullptr, nullptr, CB * LL, CC, CC, ACT_TANH, 1);
    gemm(mm, 0, CC, 0, mp1W + wofs, 0, CC, 0, mpv, 0, CC,
         mp1b + i * CC, nullptr, nullptr, CB, CC, CC, ACT_TANH, 1);
    gate_mul<<<(unsigned)((BIG + 255) / 256), 256, 0, stream>>>(hph, mpv, c2p, LL, CC, BIG);
    // attention scores + masked softmax
    vec_dot_rows<<<(CB * NN_ + 255) / 256, 256, 0, stream>>>(hch, hc1W + (long long)i * CC, hc1b + i, ssrc, (long long)CB * NN_, CC);
    mask_softmax_k<<<CB, 256, 0, stream>>>(ssrc, atoms_mask, catt, NN_);
    vec_dot_rows<<<(CB * LL + 255) / 256, 256, 0, stream>>>(hph, hp1W + (long long)i * CC, hp1b + i, sdst, (long long)CB * LL, CC);
    mask_softmax_k<<<CB, 256, 0, stream>>>(sdst, amino_mask, patt, LL);
    // weighted pooled features + GRU memory update
    att_reduce<<<CB, CC, 0, stream>>>(cfeat, catt, cf, NN_, CC);
    att_reduce<<<CB, CC, 0, stream>>>(pfeat, patt, pf, LL, CC);
    gru_step<<<CB, 3 * CC, 0, stream>>>(cf, pf, mm, gru_wih, gru_whh, gru_bih, gru_bhh, CC);
  }

  // ---------------- output head ----------------
  final_out<<<CB, 256, 0, stream>>>(cf, sfin, pf, WoutW, Woutb, (float*)d_out);
}
